// GroupedQueryAttention_15728170238370
// MI455X (gfx1250) — matrix operand order repaired and checked by reading, not compiled, not run on hardware
//
#include <hip/hip_runtime.h>

// ---------------------------------------------------------------------------
// GQA forward for MI455X (gfx1250, wave32).  All matmul work goes through
// v_wmma_f32_16x16x32_bf16 (f32 accumulate).  Compute-bound workload
// (~120 GFLOP vs ~5us of HBM traffic at 23.3 TB/s), so bf16 WMMA everywhere.
// ---------------------------------------------------------------------------

typedef __bf16 bf16;
typedef __attribute__((ext_vector_type(16))) __bf16 bf16x16;
typedef __attribute__((ext_vector_type(8)))  float  f32x8;
typedef __attribute__((ext_vector_type(4)))  float  f32x4;

__device__ __forceinline__ f32x8 wmma_bf16(bf16x16 a, bf16x16 b, f32x8 c) {
  // D = A(16x32) * B(32x16) + C(16x16), f32 accumulate
  return __builtin_amdgcn_wmma_f32_16x16x32_bf16(false, a, false, b,
                                                 (short)0, c, false, false);
}

// Load a 16-element bf16 fragment as two 16-byte chunks (ds_load_b128 pairs).
__device__ __forceinline__ bf16x16 ld_frag(const bf16* p0, const bf16* p1) {
  union { bf16x16 v; f32x4 f[2]; } u;
  u.f[0] = *reinterpret_cast<const f32x4*>(p0);
  u.f[1] = *reinterpret_cast<const f32x4*>(p1);
  return u.v;
}

// ---------------------------------------------------------------------------
// f32 -> bf16 conversion (bandwidth trivial vs compute)
// ---------------------------------------------------------------------------
__global__ void gqa_cvt_bf16(const float* __restrict__ in,
                             bf16* __restrict__ out, int n) {
  int i = blockIdx.x * blockDim.x + threadIdx.x;
  if (i < n) out[i] = (bf16)in[i];
}

// ---------------------------------------------------------------------------
// C[M,N] = A[M,K] * W[N,K]^T + bias.  Block tile 128x128x32, 256 threads
// (8 waves as 2x4); each wave owns a 64x32 tile = 4x2 WMMA accumulators.
// ---------------------------------------------------------------------------
#define GEMM_LD 40  // padded LDS k-stride (80B rows, 16B-aligned chunks)

__global__ __launch_bounds__(256)
void gqa_gemm_bf16(const bf16* __restrict__ A,     // [M,K] row-major
                   const bf16* __restrict__ W,     // [N,K] row-major
                   const float* __restrict__ bias, // [N]
                   float* __restrict__ C,          // [M,N] f32
                   int M, int N, int K) {
  __shared__ bf16 sA[128][GEMM_LD];
  __shared__ bf16 sW[128][GEMM_LD];
  const int tid  = threadIdx.x;
  const int lane = tid & 31, wid = tid >> 5;
  const int wm = wid >> 2, wn = wid & 3;    // 2 (M) x 4 (N) waves
  const int hf = lane >> 4, l16 = lane & 15;
  const int bm0 = blockIdx.y * 128, bn0 = blockIdx.x * 128;
  const int lr = tid >> 1, lc = (tid & 1) * 16;  // cooperative-load coords

  f32x8 acc[4][2] = {};

  for (int k0 = 0; k0 < K; k0 += 32) {
    const f32x4* ga = reinterpret_cast<const f32x4*>(A + (size_t)(bm0 + lr) * K + k0 + lc);
    const f32x4* gw = reinterpret_cast<const f32x4*>(W + (size_t)(bn0 + lr) * K + k0 + lc);
    f32x4 a0 = ga[0], a1 = ga[1];
    f32x4 w0 = gw[0], w1 = gw[1];
    if (k0 + 32 < K) {  // global_prefetch_b8 for next k-tile
      __builtin_prefetch(A + (size_t)(bm0 + lr) * K + k0 + 32 + lc, 0, 0);
      __builtin_prefetch(W + (size_t)(bn0 + lr) * K + k0 + 32 + lc, 0, 0);
    }
    *reinterpret_cast<f32x4*>(&sA[lr][lc])     = a0;
    *reinterpret_cast<f32x4*>(&sA[lr][lc + 8]) = a1;
    *reinterpret_cast<f32x4*>(&sW[lr][lc])     = w0;
    *reinterpret_cast<f32x4*>(&sW[lr][lc + 8]) = w1;
    __syncthreads();

    bf16x16 af[4], bfg[2];
#pragma unroll
    for (int i = 0; i < 4; ++i) {  // A frag: row = lane&15, K = {h*8.., 16+h*8..}
      const bf16* p = &sA[wm * 64 + i * 16 + l16][hf * 8];
      af[i] = ld_frag(p, p + 16);
    }
#pragma unroll
    for (int j = 0; j < 2; ++j) {  // B frag: col = lane&15, K = {h*8.., 16+h*8..}
      const bf16* p = &sW[wn * 32 + j * 16 + l16][hf * 8];
      bfg[j] = ld_frag(p, p + 16);
    }
#pragma unroll
    for (int i = 0; i < 4; ++i)
#pragma unroll
      for (int j = 0; j < 2; ++j)
        acc[i][j] = wmma_bf16(af[i], bfg[j], acc[i][j]);
    __syncthreads();
  }

  // C layout: VGPR r -> row r (lanes 0-15) / r+8 (lanes 16-31), col = lane&15
#pragma unroll
  for (int i = 0; i < 4; ++i)
#pragma unroll
    for (int j = 0; j < 2; ++j) {
      const int col = bn0 + wn * 32 + j * 16 + l16;
      const float bv = bias[col];
#pragma unroll
      for (int r = 0; r < 8; ++r) {
        const int row = bm0 + wm * 64 + i * 16 + r + hf * 8;
        C[(size_t)row * N + col] = acc[i][j][r] + bv;
      }
    }
}

// ---------------------------------------------------------------------------
// RoPE + head-major bf16 repack.  For q: SCALE * key_weights[h] folded in
// (exact, since softmax scaling is pre-softmax and per-row constant).
// src: [B*S, heads*64] f32   dst: [B*heads, S, 64] bf16
// ---------------------------------------------------------------------------
__global__ void gqa_rope_pack(const float* __restrict__ src,
                              const float* __restrict__ kw,  // [heads] or null
                              bf16* __restrict__ dst,
                              int heads, int do_rope, float base_scale, int S) {
  int idx = blockIdx.x * blockDim.x + threadIdx.x;
  int total = 2 * S * heads * 32;
  if (idx >= total) return;
  int d = idx & 31; int t = idx >> 5;
  int h = t % heads; t /= heads;
  int s = t % S;
  int b = t / S;
  const float* row = src + ((size_t)(b * S + s) * heads + h) * 64;
  float x1 = row[d], x2 = row[d + 32];
  float o1 = x1, o2 = x2;
  if (do_rope) {
    float invf = __powf(10000.f, -(float)(2 * d) * (1.0f / 64.0f));
    float ang = (float)s * invf;
    float sn, cs;
    __sincosf(ang, &sn, &cs);
    o1 = x1 * cs - x2 * sn;
    o2 = x2 * cs + x1 * sn;
  }
  float sc = base_scale * (kw ? kw[h] : 1.0f);
  o1 *= sc; o2 *= sc;
  bf16* drow = dst + ((size_t)(b * heads + h) * S + s) * 64;
  drow[d]      = (bf16)o1;
  drow[d + 32] = (bf16)o2;
}

// ---------------------------------------------------------------------------
// Causal flash attention.  Block = 4 waves x 16 q-rows = 64 q rows.
// Per key tile (64 keys): S = q*k^T via 8 WMMAs/wave, online softmax with
// 16-lane shfl_xor row reductions (matches C layout: VGPR r <-> row r/r+8),
// P re-laid out through wave-private LDS, O += P*V via 8 WMMAs/wave.
// ---------------------------------------------------------------------------
#define FA_PAD 72  // 144B rows: 16B-aligned chunks, spreads LDS banks

__global__ __launch_bounds__(128)
void gqa_flash_attn(const bf16* __restrict__ qh,  // [B*32, S, 64] pre-scaled
                    const bf16* __restrict__ kh,  // [B*8,  S, 64]
                    const bf16* __restrict__ vh,  // [B*8,  S, 64]
                    bf16* __restrict__ oat,       // [B*S, 32*64]
                    int S) {
  __shared__ bf16 skt[64][FA_PAD];    // K tile, [key][dim]
  __shared__ bf16 svt[64][FA_PAD];    // V tile transposed, [dim][key]
  __shared__ bf16 sp[4][16][FA_PAD];  // per-wave P tile, [row][key]
  const int tid  = threadIdx.x;
  const int lane = tid & 31, wid = tid >> 5;
  const int hf = lane >> 4, l16 = lane & 15;
  const int qt = blockIdx.x;
  const int head = blockIdx.y;               // b*32 + h
  const int b = head >> 5, h = head & 31, kv = h >> 2;
  const int qbase = qt * 64;
  const int qrow0 = qbase + wid * 16;

  bf16x16 aq[2];  // q A-fragments, D split into two K=32 steps
#pragma unroll
  for (int s = 0; s < 2; ++s) {
    const bf16* p = qh + ((size_t)head * S + qrow0 + l16) * 64 + s * 32 + hf * 8;
    aq[s] = ld_frag(p, p + 16);
  }

  float mrow[8], lrow[8];
  f32x8 oacc[4] = {};
#pragma unroll
  for (int r = 0; r < 8; ++r) { mrow[r] = -1e30f; lrow[r] = 0.f; }

  const bf16* kb0 = kh + (size_t)(b * 8 + kv) * S * 64;
  const bf16* vb0 = vh + (size_t)(b * 8 + kv) * S * 64;

  for (int it = 0; it <= qt; ++it) {
    const int kstart = it * 64;
    {  // cooperative tile load: 128 threads, 32 bf16 each
      const int r = tid >> 1, c = (tid & 1) * 32;
      const f32x4* gk = reinterpret_cast<const f32x4*>(kb0 + (size_t)(kstart + r) * 64 + c);
      f32x4 k0 = gk[0], k1 = gk[1], k2 = gk[2], k3 = gk[3];
      *reinterpret_cast<f32x4*>(&skt[r][c])      = k0;
      *reinterpret_cast<f32x4*>(&skt[r][c + 8])  = k1;
      *reinterpret_cast<f32x4*>(&skt[r][c + 16]) = k2;
      *reinterpret_cast<f32x4*>(&skt[r][c + 24]) = k3;
      const bf16* gv = vb0 + (size_t)(kstart + r) * 64 + c;
#pragma unroll
      for (int e = 0; e < 32; ++e) svt[c + e][r] = gv[e];  // transpose V
    }
    __syncthreads();

    // scores: 16x64 per wave (4 key tiles x 2 k-steps)
    f32x8 sc[4];
#pragma unroll
    for (int t = 0; t < 4; ++t) {
      f32x8 a = {};
#pragma unroll
      for (int s = 0; s < 2; ++s) {
        const bf16* p = &skt[t * 16 + l16][s * 32 + hf * 8];
        a = wmma_bf16(aq[s], ld_frag(p, p + 16), a);
      }
      sc[t] = a;
    }

    // causal mask + per-row running max (row = r + 8*hf within wave tile)
    float tmax[8];
#pragma unroll
    for (int r = 0; r < 8; ++r) {
      const int qi = qbase + wid * 16 + r + hf * 8;
      float mv = -1e30f;
#pragma unroll
      for (int t = 0; t < 4; ++t) {
        float v = sc[t][r];
        if (kstart + t * 16 + l16 > qi) v = -1e30f;
        sc[t][r] = v;
        mv = fmaxf(mv, v);
      }
      tmax[r] = mv;
    }
#pragma unroll
    for (int r = 0; r < 8; ++r)
#pragma unroll
      for (int off = 1; off < 16; off <<= 1)
        tmax[r] = fmaxf(tmax[r], __shfl_xor(tmax[r], off, 32));

    float fac[8], rsum[8];
#pragma unroll
    for (int r = 0; r < 8; ++r) {
      float mnew = fmaxf(mrow[r], tmax[r]);
      fac[r] = __expf(mrow[r] - mnew);
      mrow[r] = mnew;
    }
#pragma unroll
    for (int r = 0; r < 8; ++r) {
      float srow = 0.f;
#pragma unroll
      for (int t = 0; t < 4; ++t) {
        float pv = __expf(sc[t][r] - mrow[r]);
        sc[t][r] = pv;
        srow += pv;
      }
      rsum[r] = srow;
    }
#pragma unroll
    for (int r = 0; r < 8; ++r)
#pragma unroll
      for (int off = 1; off < 16; off <<= 1)
        rsum[r] += __shfl_xor(rsum[r], off, 32);
#pragma unroll
    for (int r = 0; r < 8; ++r) lrow[r] = lrow[r] * fac[r] + rsum[r];
#pragma unroll
    for (int u = 0; u < 4; ++u)
#pragma unroll
      for (int r = 0; r < 8; ++r)
        oacc[u][r] = oacc[u][r] * fac[r];

    // P: C layout -> A layout via wave-private LDS round trip
#pragma unroll
    for (int t = 0; t < 4; ++t)
#pragma unroll
      for (int r = 0; r < 8; ++r)
        sp[wid][r + hf * 8][t * 16 + l16] = (bf16)sc[t][r];
    __syncthreads();

    // O += P * V (contraction over 64 keys, two K=32 steps)
#pragma unroll
    for (int u = 0; u < 4; ++u) {
#pragma unroll
      for (int s = 0; s < 2; ++s) {
        const bf16* pa = &sp[wid][l16][s * 32 + hf * 8];
        const bf16* pb = &svt[u * 16 + l16][s * 32 + hf * 8];
        oacc[u] = wmma_bf16(ld_frag(pa, pa + 16), ld_frag(pb, pb + 16), oacc[u]);
      }
    }
    __syncthreads();
  }

  // normalize + store into [B*S, 32*64] for the O-projection GEMM
#pragma unroll
  for (int u = 0; u < 4; ++u)
#pragma unroll
    for (int r = 0; r < 8; ++r) {
      const int qi = qbase + wid * 16 + r + hf * 8;
      float o = oacc[u][r] / lrow[r];
      oat[((size_t)(b * S + qi)) * 2048 + h * 64 + u * 16 + l16] = (bf16)o;
    }
}

// ---------------------------------------------------------------------------
// Launch
// ---------------------------------------------------------------------------
extern "C" void kernel_launch(void* const* d_in, const int* in_sizes, int n_in,
                              void* d_out, int out_size, void* d_ws, size_t ws_size,
                              hipStream_t stream) {
  constexpr int Bsz = 2, S = 2048, HID = 2048, NH = 32, NKV = 8, HD = 64;
  constexpr size_t Mr = (size_t)Bsz * S;  // 4096 rows

  const float* x  = (const float*)d_in[0];
  const float* Wq = (const float*)d_in[1];
  const float* bq = (const float*)d_in[2];
  const float* Wk = (const float*)d_in[3];
  const float* bk = (const float*)d_in[4];
  const float* Wv = (const float*)d_in[5];
  const float* bv = (const float*)d_in[6];
  const float* Wo = (const float*)d_in[7];
  const float* bo = (const float*)d_in[8];
  const float* kw = (const float*)d_in[9];
  float* out = (float*)d_out;

  char* p = (char*)d_ws;
  auto alloc = [&](size_t bytes) {
    char* r = p; p += (bytes + 255) & ~(size_t)255; return r;
  };
  bf16* xb  = (bf16*)alloc(Mr * HID * 2);
  bf16* Wqb = (bf16*)alloc((size_t)NH * HD * HID * 2);
  bf16* Wkb = (bf16*)alloc((size_t)NKV * HD * HID * 2);
  bf16* Wvb = (bf16*)alloc((size_t)NKV * HD * HID * 2);
  bf16* Wob = (bf16*)alloc((size_t)HID * NH * HD * 2);
  float* qf = (float*)alloc(Mr * NH * HD * 4);
  float* kf = (float*)alloc(Mr * NKV * HD * 4);
  float* vf = (float*)alloc(Mr * NKV * HD * 4);
  bf16* qhd = (bf16*)alloc(Mr * NH * HD * 2);
  bf16* khd = (bf16*)alloc(Mr * NKV * HD * 2);
  bf16* vhd = (bf16*)alloc(Mr * NKV * HD * 2);
  bf16* oat = (bf16*)qf;  // qf scratch is dead after rope/pack; reuse (16MB<33MB)

  auto nb = [](size_t n) { return dim3((unsigned)((n + 255) / 256)); };

  gqa_cvt_bf16<<<nb(Mr * HID), 256, 0, stream>>>(x, xb, (int)(Mr * HID));
  gqa_cvt_bf16<<<nb((size_t)NH * HD * HID), 256, 0, stream>>>(Wq, Wqb, NH * HD * HID);
  gqa_cvt_bf16<<<nb((size_t)NKV * HD * HID), 256, 0, stream>>>(Wk, Wkb, NKV * HD * HID);
  gqa_cvt_bf16<<<nb((size_t)NKV * HD * HID), 256, 0, stream>>>(Wv, Wvb, NKV * HD * HID);
  gqa_cvt_bf16<<<nb((size_t)HID * NH * HD), 256, 0, stream>>>(Wo, Wob, HID * NH * HD);

  gqa_gemm_bf16<<<dim3((NH * HD) / 128, (unsigned)(Mr / 128)), 256, 0, stream>>>(
      xb, Wqb, bq, qf, (int)Mr, NH * HD, HID);
  gqa_gemm_bf16<<<dim3((NKV * HD) / 128, (unsigned)(Mr / 128)), 256, 0, stream>>>(
      xb, Wkb, bk, kf, (int)Mr, NKV * HD, HID);
  gqa_gemm_bf16<<<dim3((NKV * HD) / 128, (unsigned)(Mr / 128)), 256, 0, stream>>>(
      xb, Wvb, bv, vf, (int)Mr, NKV * HD, HID);

  // SCALE = 64^-0.5 = 0.125; key_weights folded into q
  gqa_rope_pack<<<nb((size_t)Bsz * S * NH * 32), 256, 0, stream>>>(
      qf, kw, qhd, NH, 1, 0.125f, S);
  gqa_rope_pack<<<nb((size_t)Bsz * S * NKV * 32), 256, 0, stream>>>(
      kf, nullptr, khd, NKV, 1, 1.0f, S);
  gqa_rope_pack<<<nb((size_t)Bsz * S * NKV * 32), 256, 0, stream>>>(
      vf, nullptr, vhd, NKV, 0, 1.0f, S);

  gqa_flash_attn<<<dim3(S / 64, Bsz * NH), 128, 0, stream>>>(qhd, khd, vhd, oat, S);

  gqa_gemm_bf16<<<dim3(HID / 128, (unsigned)(Mr / 128)), 256, 0, stream>>>(
      oat, Wob, bo, out, (int)Mr, HID, NH * HD);

  (void)in_sizes; (void)n_in; (void)out_size; (void)ws_size;
}
